// GCN_89094801588865
// MI455X (gfx1250) — matrix operand order repaired and checked by reading, not compiled, not run on hardware
//
#include <hip/hip_runtime.h>
#include <hip/hip_bf16.h>
#include <cstdint>

// ---------------------------------------------------------------------------
// GCN inference for MI455X (gfx1250).
//  - GEMMs: v_wmma_f32_16x16x32_bf16, fp32 accumulate, double-buffered LDS B
//    tiles filled via GLOBAL_LOAD_ASYNC_TO_LDS_B128 (ASYNCcnt) when available.
//  - SpMM:  bf16 gather (halves random-read traffic), f32 hardware atomics
//           for scatter (dst working set ~200MB ~= 192MB L2).
//  - Softmax/bias: wave32 shuffles / float4 streaming.
// Workspace layout (needs ~308 MB):
//   [Wt0 bf16 512x512][Wt1 bf16 512x512][Wt2 bf16 64x512][G bf16 M*512][S f32 M*512]
// ---------------------------------------------------------------------------

typedef __attribute__((ext_vector_type(16))) __bf16 v16bf;
typedef __attribute__((ext_vector_type(2)))  __bf16 v2bf;
typedef __attribute__((ext_vector_type(8)))  float  v8f;
typedef int v4i_ __attribute__((__vector_size__(16)));  // matches builtin param

#if defined(__gfx1250__) && __has_builtin(__builtin_amdgcn_global_load_async_to_lds_b128) && \
    __has_builtin(__builtin_amdgcn_s_wait_asynccnt)
#define USE_ASYNC_LDS 1
#else
#define USE_ASYNC_LDS 0
#endif

// Pack W[K][Nc] (fp32, row-major) -> Wt[Nc][K] (bf16, row-major) so that each
// WMMA B-fragment lane load is 16 contiguous bf16 (32B).
__global__ void pack_wt_kernel(const float* __restrict__ W, __bf16* __restrict__ Wt,
                               int K, int Nc) {
  int idx = blockIdx.x * blockDim.x + threadIdx.x;
  if (idx >= K * Nc) return;
  int n = idx / K;
  int k = idx - n * K;
  Wt[idx] = (__bf16)W[(size_t)k * Nc + n];
}

__global__ void zero_f32_kernel(float4* __restrict__ p, long n4) {
  long i = (long)blockIdx.x * blockDim.x + threadIdx.x;
  long stride = (long)gridDim.x * blockDim.x;
  float4 z = make_float4(0.f, 0.f, 0.f, 0.f);
  for (; i < n4; i += stride) p[i] = z;
}

// C[M][NOUT](bf16) = A[M][512](f32) * Wt^T   (Wt is [NOUT][512] bf16)
// Block: 256 threads = 8 waves. Block tile: 128 rows x 64 cols.
// Each wave: 16 rows x 64 cols = 4 accumulators of v_wmma_f32_16x16x32_bf16.
// B tile (64x32 bf16 = 4KB) double-buffered in LDS; next tile's fill is issued
// asynchronously and overlaps the current tile's 4 WMMAs + A-fragment loads.
template <int NOUT>
__global__ __launch_bounds__(256) void gemm_bf16_wmma(
    const float* __restrict__ A, const __bf16* __restrict__ Wt,
    __bf16* __restrict__ C, int M) {
  __shared__ alignas(32) __bf16 sB[2][64 * 32];  // 2 x 4KB

  const int lane  = threadIdx.x & 31;
  const int wave  = threadIdx.x >> 5;
  const int half  = lane >> 4;   // 0: K-low half lanes, 1: K-high half lanes
  const int l16   = lane & 15;
  const int mbase = blockIdx.x * 128 + wave * 16;
  const int n0    = blockIdx.y * 64;

  int r = mbase + l16;
  if (r >= M) r = M - 1;  // clamp for safe reads; stores are guarded

  // B staging: thread t transfers 8 bf16 (16B) of Wt[n0 + t/4][k0 + (t&3)*8 ..]
  const int scol = threadIdx.x >> 2;  // 0..63
  const int skc  = threadIdx.x & 3;   // 0..3
  const float4* __restrict__ wsrc = (const float4*)(Wt + (size_t)(n0 + scol) * 512);

  auto stage = [&](int buf, int k0) {
    const float4* src = wsrc + (k0 >> 3) + skc;
#if USE_ASYNC_LDS
    // global -> LDS without touching VGPRs; tracked by ASYNCcnt.
    __builtin_amdgcn_global_load_async_to_lds_b128(
        (__attribute__((address_space(1))) v4i_*)(uintptr_t)src,
        (__attribute__((address_space(3))) v4i_*)(uintptr_t)(&sB[buf][(size_t)threadIdx.x * 8]),
        0, 0);
#else
    *(float4*)(&sB[buf][(size_t)threadIdx.x * 8]) = *src;
#endif
  };

  v8f acc[4] = {};

  stage(0, 0);  // prologue: fill buffer 0 with K-tile 0

#pragma unroll
  for (int kt = 0; kt < 16; ++kt) {
    const int k0 = kt * 32;
#if USE_ASYNC_LDS
    __builtin_amdgcn_s_wait_asynccnt(0);  // my current-tile transfer landed
#endif
    __syncthreads();  // whole tile visible; prior reads of the other buffer done
    if (kt + 1 < 16) stage((kt + 1) & 1, k0 + 32);  // overlaps compute below

    // A fragment (contiguous K order, matching B):
    //   lanes 0-15:  row=l16, K = k0..k0+15
    //   lanes 16-31: row=l16, K = k0+16..k0+31
    const float* ar = A + (size_t)r * 512 + k0 + half * 16;
    float4 f0 = *(const float4*)(ar);
    float4 f1 = *(const float4*)(ar + 4);
    float4 f2 = *(const float4*)(ar + 8);
    float4 f3 = *(const float4*)(ar + 12);
    v16bf a;
    a[0]  = (__bf16)f0.x; a[1]  = (__bf16)f0.y; a[2]  = (__bf16)f0.z; a[3]  = (__bf16)f0.w;
    a[4]  = (__bf16)f1.x; a[5]  = (__bf16)f1.y; a[6]  = (__bf16)f1.z; a[7]  = (__bf16)f1.w;
    a[8]  = (__bf16)f2.x; a[9]  = (__bf16)f2.y; a[10] = (__bf16)f2.z; a[11] = (__bf16)f2.w;
    a[12] = (__bf16)f3.x; a[13] = (__bf16)f3.y; a[14] = (__bf16)f3.z; a[15] = (__bf16)f3.w;

    const __bf16* bbase = &sB[kt & 1][0];
#pragma unroll
    for (int nt = 0; nt < 4; ++nt) {
      // B fragment: lane = column (nt*16 + l16); lanes 0-15 hold K 0..15,
      // lanes 16-31 hold K 16..31 -> 32B contiguous in the staged tile.
      v16bf b = *(const v16bf*)(bbase + (size_t)(nt * 16 + l16) * 32 + half * 16);
      acc[nt] = __builtin_amdgcn_wmma_f32_16x16x32_bf16(
          /*neg_a=*/false, a, /*neg_b=*/false, b,
          /*c_mod=*/(short)0, acc[nt], /*reuse_a=*/false, /*reuse_b=*/false);
    }
  }

  // C/D layout: VGPR i -> row = mbase + 8*half + i, col = lane%16.
  // NOUT is compile-time: row stride folds into immediate store offsets.
  const bool full = (mbase + 16) <= M;
#pragma unroll
  for (int nt = 0; nt < 4; ++nt) {
    const int col = n0 + nt * 16 + l16;
    __bf16* cp = C + (size_t)(mbase + half * 8) * NOUT + col;
    if (full) {
#pragma unroll
      for (int i = 0; i < 8; ++i) cp[(size_t)i * NOUT] = (__bf16)acc[nt][i];
    } else {
#pragma unroll
      for (int i = 0; i < 8; ++i)
        if (mbase + half * 8 + i < M) cp[(size_t)i * NOUT] = (__bf16)acc[nt][i];
    }
  }
}

// out[dst] += w * G[src]  for 512 bf16 features; one wave32 per edge.
__global__ __launch_bounds__(256) void spmm512_bf16(
    const int* __restrict__ src, const int* __restrict__ dst,
    const float* __restrict__ ew, const __bf16* __restrict__ G,
    float* __restrict__ out, int E) {
  const int lane = threadIdx.x & 31;
  long w  = ((long)blockIdx.x * blockDim.x + threadIdx.x) >> 5;
  long nw = ((long)gridDim.x * blockDim.x) >> 5;
  for (long e = w; e < E; e += nw) {
    int   s  = src[e];
    int   d  = dst[e];
    float wt = ew[e];
    if (e + nw < E)  // hide latency of the next random 1KB row
      __builtin_prefetch(G + (size_t)src[e + nw] * 512, 0, 1);
    v16bf g = *(const v16bf*)(G + (size_t)s * 512 + lane * 16);  // 32B/lane
    float* o = out + (size_t)d * 512 + lane * 16;
#pragma unroll
    for (int i = 0; i < 16; ++i)
      unsafeAtomicAdd(o + i, wt * (float)g[i]);  // global_atomic_add_f32
  }
}

// Same for 64 features (layer 3): 2 bf16 per lane.
__global__ __launch_bounds__(256) void spmm64_bf16(
    const int* __restrict__ src, const int* __restrict__ dst,
    const float* __restrict__ ew, const __bf16* __restrict__ G,
    float* __restrict__ out, int E) {
  const int lane = threadIdx.x & 31;
  long w  = ((long)blockIdx.x * blockDim.x + threadIdx.x) >> 5;
  long nw = ((long)gridDim.x * blockDim.x) >> 5;
  for (long e = w; e < E; e += nw) {
    int   s  = src[e];
    int   d  = dst[e];
    float wt = ew[e];
    v2bf g = *(const v2bf*)(G + (size_t)s * 64 + lane * 2);
    float* o = out + (size_t)d * 64 + lane * 2;
    unsafeAtomicAdd(o + 0, wt * (float)g[0]);
    unsafeAtomicAdd(o + 1, wt * (float)g[1]);
  }
}

// S[i] = relu(S[i] + b[i % 512]); float4 streaming (128 float4 per row).
__global__ void bias_relu_kernel(float* __restrict__ S, const float* __restrict__ b,
                                 long n4) {
  long i = (long)blockIdx.x * blockDim.x + threadIdx.x;
  long stride = (long)gridDim.x * blockDim.x;
  float4* S4 = (float4*)S;
  const float4* b4 = (const float4*)b;
  for (; i < n4; i += stride) {
    float4 v  = S4[i];
    float4 bb = b4[i & 127];
    v.x = fmaxf(v.x + bb.x, 0.f);
    v.y = fmaxf(v.y + bb.y, 0.f);
    v.z = fmaxf(v.z + bb.z, 0.f);
    v.w = fmaxf(v.w + bb.w, 0.f);
    S4[i] = v;
  }
}

// In-place softmax over 64 classes; one wave32 per row, 2 values per lane.
__global__ __launch_bounds__(256) void softmax64_kernel(
    float* __restrict__ S, const float* __restrict__ b, int M) {
  const int lane = threadIdx.x & 31;
  long row = ((long)blockIdx.x * blockDim.x + threadIdx.x) >> 5;
  if (row >= M) return;
  float2 v = *(float2*)(S + (size_t)row * 64 + lane * 2);
  v.x += b[lane * 2];
  v.y += b[lane * 2 + 1];
  float m = fmaxf(v.x, v.y);
#pragma unroll
  for (int off = 16; off > 0; off >>= 1) m = fmaxf(m, __shfl_xor(m, off, 32));
  float e0 = __expf(v.x - m);
  float e1 = __expf(v.y - m);
  float s = e0 + e1;
#pragma unroll
  for (int off = 16; off > 0; off >>= 1) s += __shfl_xor(s, off, 32);
  float inv = 1.0f / s;
  *(float2*)(S + (size_t)row * 64 + lane * 2) = make_float2(e0 * inv, e1 * inv);
}

extern "C" void kernel_launch(void* const* d_in, const int* in_sizes, int n_in,
                              void* d_out, int out_size, void* d_ws, size_t ws_size,
                              hipStream_t stream) {
  const float* X    = (const float*)d_in[0];
  const int*   esrc = (const int*)d_in[1];
  const int*   edst = (const int*)d_in[2];
  const float* ew   = (const float*)d_in[3];
  const float* W0   = (const float*)d_in[4];
  const float* b0   = (const float*)d_in[5];
  const float* W1   = (const float*)d_in[6];
  const float* b1   = (const float*)d_in[7];
  const float* W2   = (const float*)d_in[8];
  const float* b2   = (const float*)d_in[9];

  const int DIN = 512, HID = 512, NC = 64;
  const int M = in_sizes[0] / DIN;  // 100000 nodes
  const int E = in_sizes[1];        // 3200000 edges

  // Workspace carve-up (all offsets 256B-aligned).
  __bf16* Wt0 = (__bf16*)d_ws;
  __bf16* Wt1 = Wt0 + (size_t)DIN * HID;
  __bf16* Wt2 = Wt1 + (size_t)HID * HID;
  __bf16* G   = Wt2 + (size_t)HID * NC;           // bf16 [M][512] gather source
  float*  S   = (float*)(G + (size_t)M * HID);    // f32  [M][512] scatter target
  float*  out = (float*)d_out;                    // f32  [M][64]

  dim3 blk(256);
  dim3 g512((M + 127) / 128, HID / 64);
  dim3 g64((M + 127) / 128, 1);

  // Pack weights once per launch (tiny).
  pack_wt_kernel<<<(DIN * HID + 255) / 256, blk, 0, stream>>>(W0, Wt0, DIN, HID);
  pack_wt_kernel<<<(HID * HID + 255) / 256, blk, 0, stream>>>(W1, Wt1, HID, HID);
  pack_wt_kernel<<<(HID * NC + 255) / 256, blk, 0, stream>>>(W2, Wt2, HID, NC);

  // Layer 1: H1 = relu(spmm(X @ W0) + b0)
  gemm_bf16_wmma<512><<<g512, blk, 0, stream>>>(X, Wt0, G, M);
  zero_f32_kernel<<<8192, blk, 0, stream>>>((float4*)S, (long)M * HID / 4);
  spmm512_bf16<<<4096, blk, 0, stream>>>(esrc, edst, ew, G, S, E);
  bias_relu_kernel<<<8192, blk, 0, stream>>>(S, b0, (long)M * HID / 4);

  // Layer 2: H2 = relu(spmm(H1 @ W1) + b1)
  gemm_bf16_wmma<512><<<g512, blk, 0, stream>>>(S, Wt1, G, M);
  zero_f32_kernel<<<8192, blk, 0, stream>>>((float4*)S, (long)M * HID / 4);
  spmm512_bf16<<<4096, blk, 0, stream>>>(esrc, edst, ew, G, S, E);
  bias_relu_kernel<<<8192, blk, 0, stream>>>(S, b1, (long)M * HID / 4);

  // Layer 3: out = softmax(spmm(H2 @ W2) + b2)
  gemm_bf16_wmma<64><<<g64, blk, 0, stream>>>(S, Wt2, G, M);
  zero_f32_kernel<<<2048, blk, 0, stream>>>((float4*)out, (long)M * NC / 4);
  spmm64_bf16<<<4096, blk, 0, stream>>>(esrc, edst, ew, G, out, E);
  softmax64_kernel<<<(M + 7) / 8, blk, 0, stream>>>(out, b2, M);
}
